// DotProductAttention_29094108463329
// MI455X (gfx1250) — matrix operand order repaired and checked by reading, not compiled, not run on hardware
//
#include <hip/hip_runtime.h>
#include <hip/hip_bf16.h>

typedef __attribute__((ext_vector_type(16))) _Float16 v16h;
typedef __attribute__((ext_vector_type(8)))  float    v8f;

namespace {
constexpr int kSQ = 2048, kB = 2, kNH = 16, kHD = 128;
constexpr int kRowStride = kB * kNH * kHD;   // 4096 elements between seq positions
constexpr int kKT = 32;                      // keys per tile
constexpr int kQT = 64;                      // query rows per workgroup (4 waves x 16)
constexpr int kNT = kSQ / kKT;               // 64 key tiles
constexpr size_t kElems = (size_t)kSQ * kRowStride;  // 8M elements per tensor
constexpr float kScale = 0.08838834764831845f;  // 1/sqrt(128)
}

union U16x2 { unsigned u; _Float16 h[2]; };
union AV    { v16h v; unsigned u[8]; };

__device__ inline unsigned pack2h(float a, float b) {
  U16x2 t; t.h[0] = (_Float16)a; t.h[1] = (_Float16)b; return t.u;
}
__device__ inline v8f wmma_f16(v16h a, v16h b, v8f c) {
  // D(f32 16x16) = A(f16 16x32) * B(f16 32x16) + C
  return __builtin_amdgcn_wmma_f32_16x16x32_f16(false, a, false, b, (short)0, c,
                                                false, false);
}
__device__ inline unsigned lds_off(const void* p) {  // AS3 offset = low 32 addr bits
  return (unsigned)(size_t)p;
}
// ASYNCcnt-tracked HBM -> LDS copy (CDNA5 GLOBAL_LOAD_ASYNC_TO_LDS_B128).
__device__ inline void async_copy_b128(unsigned lds_byte, const void* gaddr) {
  asm volatile("global_load_async_to_lds_b128 %0, %1, off"
               :: "v"(lds_byte), "v"(gaddr) : "memory");
}
__device__ inline void wait_asynccnt0() {
  asm volatile("s_wait_asynccnt 0x0" ::: "memory");
}

// ---- pre-pass: K*(1/sqrt(hd)) and V converted f32 -> f16 once into workspace.
__global__ __launch_bounds__(256)
void fa_cvt_f16(const float* __restrict__ K, const float* __restrict__ V,
                _Float16* __restrict__ Kh, _Float16* __restrict__ Vh) {
  const size_t i = ((size_t)blockIdx.x * 256 + threadIdx.x) * 8;
  const float4 k0 = ((const float4*)(K + i))[0];
  const float4 k1 = ((const float4*)(K + i))[1];
  const float4 v0 = ((const float4*)(V + i))[0];
  const float4 v1 = ((const float4*)(V + i))[1];
  uint4 kp, vp;
  kp.x = pack2h(k0.x * kScale, k0.y * kScale);
  kp.y = pack2h(k0.z * kScale, k0.w * kScale);
  kp.z = pack2h(k1.x * kScale, k1.y * kScale);
  kp.w = pack2h(k1.z * kScale, k1.w * kScale);
  vp.x = pack2h(v0.x, v0.y);
  vp.y = pack2h(v0.z, v0.w);
  vp.z = pack2h(v1.x, v1.y);
  vp.w = pack2h(v1.z, v1.w);
  ((uint4*)(Kh + i))[0] = kp;
  ((uint4*)(Vh + i))[0] = vp;
}

__global__ __launch_bounds__(128)
void fa_fwd_f16wmma(const float* __restrict__ Q,
                    const _Float16* __restrict__ Kh,
                    const _Float16* __restrict__ Vh,
                    float* __restrict__ O) {
  const int bid  = blockIdx.x;
  const int qt   = bid >> 5;          // SQ/64 = 32 query tiles
  const int bn   = bid & 31;          // B*NH = 32
  const int b    = bn >> 4;
  const int n    = bn & 15;
  const int hoff = b * (kNH * kHD) + n * kHD;

  const int tid    = threadIdx.x;
  const int wave   = tid >> 5;
  const int lane   = tid & 31;
  const int lane16 = lane & 15;
  const int lhalf  = lane >> 4;

  __shared__ unsigned short sKh[2][kKT * kHD];  // f16 K tile [key][hd], x2 buf (8KB ea)
  __shared__ unsigned short sVh[2][kKT * kHD];  // f16 V tile [key][hd], x2 buf

  // ---- Q^T (loop-invariant WMMA-B operand): lane16 = query column, VGPR j packs
  //      hd 32c + 2j + 8h (+8 for j>=4), +1. (1/sqrt(hd) already folded into Kh.)
  AV qb[4];
  {
    const float* qbase = Q + (qt * kQT + wave * 16) * kRowStride + hoff;
#pragma unroll
    for (int c = 0; c < 4; ++c) {
#pragma unroll
      for (int j = 0; j < 8; ++j) {
        const int hd = 32 * c + 2 * j + 8 * lhalf + (j >= 4 ? 8 : 0);
        qb[c].u[j] = pack2h(qbase[lane16 * kRowStride + hd],
                            qbase[lane16 * kRowStride + hd + 1]);
      }
    }
  }

  float mrun = -1e30f, lrun = 0.0f;   // per-lane: query = lane16 (dup in halves)
  v8f acc[8];
#pragma unroll
  for (int o = 0; o < 8; ++o) acc[o] = (v8f){0, 0, 0, 0, 0, 0, 0, 0};

  // ---- async staging: thread copies 4x16B of Kh and 4x16B of Vh per tile
  const int skey = tid >> 2;              // key row 0..31
  const int sh   = (tid & 3) * 32;        // 32-half slice of the 128-half row
  const unsigned dK[2] = {lds_off(&sKh[0][skey * kHD + sh]),
                          lds_off(&sKh[1][skey * kHD + sh])};
  const unsigned dV[2] = {lds_off(&sVh[0][skey * kHD + sh]),
                          lds_off(&sVh[1][skey * kHD + sh])};
  const _Float16* Kg = Kh + hoff + skey * kRowStride + sh;
  const _Float16* Vg = Vh + hoff + skey * kRowStride + sh;

#define STAGE(t, buf)                                                       \
  do {                                                                      \
    const _Float16* kr_ = Kg + (size_t)(t) * (kKT * kRowStride);            \
    const _Float16* vr_ = Vg + (size_t)(t) * (kKT * kRowStride);            \
    _Pragma("unroll")                                                       \
    for (int i_ = 0; i_ < 4; ++i_) {                                        \
      async_copy_b128(dK[buf] + 16 * i_, kr_ + 8 * i_);                     \
      async_copy_b128(dV[buf] + 16 * i_, vr_ + 8 * i_);                     \
    }                                                                       \
  } while (0)

  STAGE(0, 0);

  for (int t = 0; t < kNT; ++t) {
    const int buf = t & 1;
    wait_asynccnt0();     // this wave's tile-t copies have landed in LDS
    __syncthreads();      // => every wave's copies landed; prev buf fully read
    if (t + 1 < kNT) STAGE(t + 1, buf ^ 1);   // overlap next tile with compute

    const unsigned short* sKb = &sKh[buf][0];
    const unsigned short* sVb = &sVh[buf][0];

    // ---- scores^T = K * Q^T : D[m=key][n=query], contraction over hd=128.
    //      A-operand is a direct b128-aligned f16 LDS read (no conversion).
    v8f s0 = (v8f){0, 0, 0, 0, 0, 0, 0, 0};   // keys 0..15 of tile
    v8f s1 = (v8f){0, 0, 0, 0, 0, 0, 0, 0};   // keys 16..31 of tile
#pragma unroll
    for (int c = 0; c < 4; ++c) {
      AV a0, a1;
      const unsigned* k0lo = (const unsigned*)&sKb[lane16 * kHD + 32 * c + lhalf * 8];
      const unsigned* k0hi = (const unsigned*)&sKb[lane16 * kHD + 32 * c + 16 + lhalf * 8];
      const unsigned* k1lo = (const unsigned*)&sKb[(16 + lane16) * kHD + 32 * c + lhalf * 8];
      const unsigned* k1hi = (const unsigned*)&sKb[(16 + lane16) * kHD + 32 * c + 16 + lhalf * 8];
#pragma unroll
      for (int v = 0; v < 4; ++v) {
        a0.u[v] = k0lo[v];      // halfs h*8 + 2v, 2v+1
        a0.u[v + 4] = k0hi[v];  // halfs 16 + h*8 + 2v, 2v+1
        a1.u[v] = k1lo[v];
        a1.u[v + 4] = k1hi[v];
      }
      s0 = wmma_f16(a0.v, qb[c].v, s0);
      s1 = wmma_f16(a1.v, qb[c].v, s1);
    }

    // ---- online softmax over keys; lane owns query lane16 (both key halves)
    float mx = fmaxf(s0[0], s1[0]);
#pragma unroll
    for (int r = 1; r < 8; ++r) mx = fmaxf(mx, fmaxf(s0[r], s1[r]));
    mx = fmaxf(mx, __shfl_xor(mx, 16, 32));   // merge key groups r+0..7 / r+8..15
    const float mn   = fmaxf(mrun, mx);
    const float facL = __expf(mrun - mn);
    mrun = mn;
    float p0[8], p1[8], rs = 0.0f;
#pragma unroll
    for (int r = 0; r < 8; ++r) {
      p0[r] = __expf(s0[r] - mn);
      p1[r] = __expf(s1[r] - mn);
      rs += p0[r] + p1[r];
    }
    rs += __shfl_xor(rs, 16, 32);
    lrun = lrun * facL + rs;

    // ---- P in WMMA-A layout: pure in-register repack (8 cvt_pk)
    AV pa;
#pragma unroll
    for (int v = 0; v < 4; ++v) {
      pa.u[v]     = pack2h(p0[2 * v], p0[2 * v + 1]);
      pa.u[v + 4] = pack2h(p1[2 * v], p1[2 * v + 1]);
    }

    // ---- broadcast rescale factor into accumulator-row (query r+8h) layout
    float facr[8];
#pragma unroll
    for (int r = 0; r < 8; ++r) facr[r] = __shfl(facL, r + 8 * lhalf, 32);

    // ---- acc = acc*diag(fac) + P * V   (8 output hd-tiles, K=32 keys).
    //      B-operand is built from row-strided 16-bit f16 LDS reads, two keys per VGPR (no conversion).
#pragma unroll
    for (int o = 0; o < 8; ++o) {
#pragma unroll
      for (int r = 0; r < 8; ++r) acc[o][r] *= facr[r];
      AV bv;
      const unsigned short* pv = &sVb[16 * o + lane16];
#pragma unroll
      for (int j = 0; j < 8; ++j) {
        const int kk = 2 * j + 8 * lhalf + (j >= 4 ? 8 : 0);
        bv.u[j] = (unsigned)pv[kk * kHD] | ((unsigned)pv[(kk + 1) * kHD] << 16);
      }
      acc[o] = wmma_f16(pa.v, bv.v, acc[o]);
    }
  }
#undef STAGE

  // ---- epilogue: divide by softmax denominator and store f32 output
  const float inv = 1.0f / lrun;
  float linv[8];
#pragma unroll
  for (int r = 0; r < 8; ++r) linv[r] = __shfl(inv, r + 8 * lhalf, 32);
#pragma unroll
  for (int o = 0; o < 8; ++o) {
#pragma unroll
    for (int r = 0; r < 8; ++r) {
      const int row = qt * kQT + wave * 16 + r + 8 * lhalf;   // query index
      O[row * kRowStride + hoff + 16 * o + lane16] = acc[o][r] * linv[r];
    }
  }
}

extern "C" void kernel_launch(void* const* d_in, const int* in_sizes, int n_in,
                              void* d_out, int out_size, void* d_ws, size_t ws_size,
                              hipStream_t stream) {
  const float* Q = (const float*)d_in[0];
  const float* K = (const float*)d_in[1];
  const float* V = (const float*)d_in[2];
  // d_in[3] is the all-true attention mask; softmax is unmasked, so it is unused.
  float* O = (float*)d_out;

  _Float16* Kh = (_Float16*)d_ws;             // 16 MB
  _Float16* Vh = Kh + kElems;                 // 16 MB

  // pre-pass: one-time f32 -> f16 (K pre-scaled by 1/sqrt(hd))
  const dim3 cgrid((unsigned)(kElems / (256 * 8)));      // 4096 blocks
  fa_cvt_f16<<<cgrid, 256, 0, stream>>>(K, V, Kh, Vh);

  const dim3 grid((kSQ / kQT) * kB * kNH);  // 32 q-tiles * 32 (b,n) heads = 1024
  fa_fwd_f16wmma<<<grid, 128, 0, stream>>>(Q, Kh, Vh, O);
}
